// MusicDecoder_24000277250352
// MI455X (gfx1250) — matrix operand order repaired and checked by reading, not compiled, not run on hardware
//
#include <hip/hip_runtime.h>
#include <hip/hip_bf16.h>
#include <math.h>

// MusicVAE-style 2-layer projected-LSTM decoder for MI455X (gfx1250).
// GEMMs on v_wmma_f32_16x16x32_bf16 (f32 accumulate); B-operand staged to LDS
// via GLOBAL_LOAD_ASYNC_TO_LDS_B128 (ASYNCcnt) with double buffering.
// Workspace requirement: ~292 MB.

typedef __attribute__((ext_vector_type(16))) __bf16 v16bf;
typedef __attribute__((ext_vector_type(8)))  __bf16 v8bf;
typedef __attribute__((ext_vector_type(8)))  float  v8f;
typedef __attribute__((ext_vector_type(4)))  int    v4i;

#if __has_builtin(__builtin_amdgcn_global_load_async_to_lds_b128) && \
    __has_builtin(__builtin_amdgcn_s_wait_asynccnt)
#define USE_ASYNC_LDS 1
#else
#define USE_ASYNC_LDS 0
#endif

__device__ __forceinline__ float sigf(float x) { return 1.0f / (1.0f + __expf(-x)); }

// Copy 16 bytes global -> LDS (async DMA path, tracked by ASYNCcnt).
__device__ __forceinline__ void stage_piece(const __bf16* g, __bf16* l) {
#if USE_ASYNC_LDS
  __builtin_amdgcn_global_load_async_to_lds_b128(
      (__attribute__((address_space(1))) v4i*)g,
      (__attribute__((address_space(3))) v4i*)l, 0, 0);
#else
  *(v8bf*)l = *(const v8bf*)g;
#endif
}
__device__ __forceinline__ void wait_async_le2() {
#if USE_ASYNC_LDS
  __builtin_amdgcn_s_wait_asynccnt(2);
#endif
}
__device__ __forceinline__ void wait_async_le0() {
#if USE_ASYNC_LDS
  __builtin_amdgcn_s_wait_asynccnt(0);
#endif
}

// ---------------- pack / convert kernels ----------------

__global__ void k_f32_to_bf16(const float* __restrict__ s, __bf16* __restrict__ d, size_t n) {
  size_t i = (size_t)blockIdx.x * blockDim.x + threadIdx.x;
  if (i < n) d[i] = (__bf16)s[i];
}

// Pack a weight matrix ([out_features, K] row-major fp32) into per-lane WMMA B-fragment bf16 (contiguous k order, as A):
//   entry = ((kb * (Ntot/16) + nb) * 32 + lane) * 16 + i
//   lane<16 : col nb*16+lane,     k = kb*32 + i        (K 0..15 half)
//   lane>=16: col nb*16+lane-16,  k = kb*32 + 16 + i   (K 16..31 half)
// gate!=0: interleave the 4H gate columns in 16-wide groups [i|f|g|o] per 16 hidden units.
__global__ void k_pack_b(const float* __restrict__ src, __bf16* __restrict__ dst,
                         int Ksrc, int kbase, int Ntot, int gate) {
  size_t tid = (size_t)blockIdx.x * blockDim.x + threadIdx.x;
  size_t total = (size_t)Ksrc * Ntot;
  if (tid >= total) return;
  int kl   = (int)(tid / Ntot);
  int ncol = (int)(tid % Ntot);
  int kg = kbase + kl;
  int origrow;
  if (gate) {
    int g16 = ncol >> 6, q = (ncol >> 4) & 3, r = ncol & 15;
    origrow = q * 2048 + g16 * 16 + r;
  } else {
    origrow = ncol;
  }
  float v = src[(size_t)origrow * Ksrc + kl];
  int kb   = kg >> 5;
  int lane = (ncol & 15) | (((kg >> 4) & 1) << 4);
  int ii   = kg & 15;
  int nb   = ncol >> 4;
  size_t di = (((size_t)kb * (size_t)(Ntot >> 4) + nb) * 32 + lane) * 16 + ii;
  dst[di] = (__bf16)v;
}

__global__ void k_pack_bias(const float* __restrict__ bih, const float* __restrict__ bhh,
                            float* __restrict__ dst) {
  int c = blockIdx.x * blockDim.x + threadIdx.x; // 0..8191
  int g16 = c >> 6, q = (c >> 4) & 3, r = c & 15;
  int row = q * 2048 + g16 * 16 + r;
  dst[c] = bih[row] + bhh[row];
}

// ---------------- fused WMMA GEMM kernels ----------------
// Block = 256 threads = 8 waves.  Wave tile = 32(M) x 64(N) -> 8 accs.
// Block tile = 256 x 64.  W staged to LDS (double-buffered 64-deep K stages).
// MODE 0: gates GEMM + LSTM pointwise.  MODE 1: projection store.
template <int MODE>
__global__ __launch_bounds__(256) void k_lstm_wmma(
    const __bf16* __restrict__ A, int lda, int Ka,     // Ka % 64 == 0
    const __bf16* __restrict__ W, int nb16,            // nb16 = Ntot/16
    const float* __restrict__ bias,                    // MODE 0 (interleaved order)
    float* __restrict__ cbuf, __bf16* __restrict__ pbuf,     // MODE 0: N x 2048
    __bf16* __restrict__ hcat, int hoff,               // MODE 1: N x 512 (+col offset)
    __bf16* __restrict__ hseq)                         // MODE 1: optional N x 256 slice
{
  __shared__ __align__(16) __bf16 ldsW[2][2 * 2048];   // 2 bufs x (2 kb-chunks x 4KB)

  const int tid   = threadIdx.x;
  const int lane  = tid & 31;
  const int wid   = tid >> 5;
  const int m0    = blockIdx.y * 256 + wid * 32;
  const int n0    = blockIdx.x * 64;
  const int rowL  = lane & 15;
  const int khalf = lane >> 4;

  v8f acc[2][4];
#pragma unroll
  for (int rg = 0; rg < 2; ++rg)
#pragma unroll
    for (int q = 0; q < 4; ++q) acc[rg][q] = (v8f){};

  // A-fragment addressing in this code's contiguous k order (matches packed W):
  // lane<16: K 0..15 ; lane>=16: K 16..31 of each 32-wide K block.
  const __bf16* arow0 = A + (size_t)(m0 + rowL) * lda + khalf * 16;
  const __bf16* arow1 = A + (size_t)(m0 + 16 + rowL) * lda + khalf * 16;

  const int nstage = Ka >> 6;             // 64 K per stage (2 WMMA k-blocks)
  const size_t chunkStride = (size_t)nb16 * 512;   // elements between consecutive kb chunks

  auto stage = [&](int s, int buf) {
    const __bf16* src = W + ((size_t)(s * 2) * nb16 + (n0 >> 4)) * 512 + tid * 8;
    stage_piece(src,               &ldsW[buf][tid * 8]);
    stage_piece(src + chunkStride, &ldsW[buf][2048 + tid * 8]);
  };

  stage(0, 0);
  for (int s = 0; s < nstage; ++s) {
    const int buf = s & 1;
    if (s + 1 < nstage) { stage(s + 1, buf ^ 1); wait_async_le2(); }
    else                { wait_async_le0(); }
    __syncthreads();                       // stage s data visible to all waves

#pragma unroll
    for (int kbl = 0; kbl < 2; ++kbl) {
      const int kb = s * 2 + kbl;
      v16bf a0, a1;
      {
        v8bf lo = *(const v8bf*)(arow0 + kb * 32);
        v8bf hi = *(const v8bf*)(arow0 + kb * 32 + 8);
        a0 = __builtin_shufflevector(lo, hi, 0,1,2,3,4,5,6,7,8,9,10,11,12,13,14,15);
      }
      {
        v8bf lo = *(const v8bf*)(arow1 + kb * 32);
        v8bf hi = *(const v8bf*)(arow1 + kb * 32 + 8);
        a1 = __builtin_shufflevector(lo, hi, 0,1,2,3,4,5,6,7,8,9,10,11,12,13,14,15);
      }
#pragma unroll
      for (int q = 0; q < 4; ++q) {
        v16bf b = *(const v16bf*)&ldsW[buf][kbl * 2048 + (q * 32 + lane) * 16];
        acc[0][q] = __builtin_amdgcn_wmma_f32_16x16x32_bf16(false, a0, false, b, (short)0, acc[0][q], false, false);
        acc[1][q] = __builtin_amdgcn_wmma_f32_16x16x32_bf16(false, a1, false, b, (short)0, acc[1][q], false, false);
      }
    }
    __syncthreads();                       // protect buf reuse (WAR vs stage s+2)
  }

  if (MODE == 0) {
    // Gate interleave: acc[rg][0..3] = i/f/g/o for the SAME 16 hidden units.
    // C/D layout: VGPR v, lane<16 -> row +v, lane>=16 -> row +8+v; col = lane&15.
    const int j  = blockIdx.x * 16 + rowL;     // hidden unit (0..2047)
    const float bi  = bias[n0 + rowL];
    const float bf_ = bias[n0 + 16 + rowL];
    const float bg  = bias[n0 + 32 + rowL];
    const float bo  = bias[n0 + 48 + rowL];
#pragma unroll
    for (int rg = 0; rg < 2; ++rg) {
#pragma unroll
      for (int v = 0; v < 8; ++v) {
        const int row = m0 + rg * 16 + v + khalf * 8;
        const size_t idx = (size_t)row * 2048 + j;
        float ig = sigf(acc[rg][0][v] + bi);
        float fg = sigf(acc[rg][1][v] + bf_);
        float gg = tanhf(acc[rg][2][v] + bg);
        float og = sigf(acc[rg][3][v] + bo);
        float c  = fg * cbuf[idx] + ig * gg;
        cbuf[idx] = c;
        pbuf[idx] = (__bf16)(og * tanhf(c));
      }
    }
  } else {
#pragma unroll
    for (int rg = 0; rg < 2; ++rg) {
#pragma unroll
      for (int q = 0; q < 4; ++q) {
        const int col = n0 + q * 16 + rowL;
#pragma unroll
        for (int v = 0; v < 8; ++v) {
          const int row = m0 + rg * 16 + v + khalf * 8;
          __bf16 hv = (__bf16)acc[rg][q][v];
          hcat[(size_t)row * 512 + hoff + col] = hv;
          if (hseq) hseq[(size_t)row * 256 + col] = hv;
        }
      }
    }
  }
}

// ---------------- softmax (wave per 256-wide row) ----------------
__global__ __launch_bounds__(256) void k_softmax(const __bf16* __restrict__ hseq,
                                                 float* __restrict__ out) {
  const int lane = threadIdx.x & 31;
  const int wid  = threadIdx.x >> 5;
  const int r = blockIdx.x * 8 + wid;   // 0 .. 16*8192-1
  const int t = r >> 13;                // step 0..15
  const int n = r & 8191;               // sequence = b*16 + bar
  const int b = n >> 4, bar = n & 15;
  const __bf16* src = hseq + ((size_t)t * 8192 + n) * 256 + lane * 8;
  v8bf hv = *(const v8bf*)src;
  float e[8];
  float mx = -3.0e38f;
#pragma unroll
  for (int i = 0; i < 8; ++i) { e[i] = (float)hv[i]; mx = fmaxf(mx, e[i]); }
#pragma unroll
  for (int d = 16; d > 0; d >>= 1) mx = fmaxf(mx, __shfl_xor(mx, d, 32));
  float sum = 0.f;
#pragma unroll
  for (int i = 0; i < 8; ++i) { e[i] = __expf(e[i] - mx); sum += e[i]; }
#pragma unroll
  for (int d = 16; d > 0; d >>= 1) sum += __shfl_xor(sum, d, 32);
  const float inv = 1.0f / sum;
  float* dst = out + ((size_t)b * 256 + bar * 16 + t) * 256 + lane * 8;
#pragma unroll
  for (int i = 0; i < 8; ++i) dst[i] = e[i] * inv;
}

// ---------------- host ----------------
extern "C" void kernel_launch(void* const* d_in, const int* in_sizes, int n_in,
                              void* d_out, int out_size, void* d_ws, size_t ws_size,
                              hipStream_t stream) {
  (void)in_sizes; (void)n_in; (void)out_size; (void)ws_size;
  const float* x    = (const float*)d_in[0];
  const float* Wih0 = (const float*)d_in[1];
  const float* Whh0 = (const float*)d_in[2];
  const float* bih0 = (const float*)d_in[3];
  const float* bhh0 = (const float*)d_in[4];
  const float* Whr0 = (const float*)d_in[5];
  const float* Wih1 = (const float*)d_in[6];
  const float* Whh1 = (const float*)d_in[7];
  const float* bih1 = (const float*)d_in[8];
  const float* bhh1 = (const float*)d_in[9];
  const float* Whr1 = (const float*)d_in[10];

  const size_t N = 8192, H = 2048, P = 256, IN = 1024, G = 8192; // G = 4H
  char* ws = (char*)d_ws;
  size_t off = 0;
  auto take = [&](size_t bytes) { size_t r = off; off += (bytes + 255) & ~(size_t)255; return r; };
  __bf16* W0ihF = (__bf16*)(ws + take(G * IN * 2));       // x -> gates0 weights (t=0)
  __bf16* W0hhF = (__bf16*)(ws + take(G * P * 2));        // h0 -> gates0
  __bf16* W1F   = (__bf16*)(ws + take(G * (2 * P) * 2));  // [h0|h1] -> gates1
  __bf16* Whr0F = (__bf16*)(ws + take(H * P * 2));
  __bf16* Whr1F = (__bf16*)(ws + take(H * P * 2));
  float*  b0    = (float*)(ws + take(G * 4));
  float*  b1    = (float*)(ws + take(G * 4));
  __bf16* xbf   = (__bf16*)(ws + take(N * IN * 2));
  __bf16* hcat  = (__bf16*)(ws + take(N * 2 * P * 2));    // [h0 | h1] per row
  float*  c0    = (float*)(ws + take(N * H * 4));
  float*  c1    = (float*)(ws + take(N * H * 4));
  __bf16* pbuf  = (__bf16*)(ws + take(N * H * 2));
  __bf16* hseq  = (__bf16*)(ws + take((size_t)16 * N * P * 2));

  const dim3 blk(256);

  // --- packing (re-done every call; deterministic, capture-safe) ---
  {
    size_t n = N * IN;
    k_f32_to_bf16<<<dim3((unsigned)((n + 255) / 256)), blk, 0, stream>>>(x, xbf, n);
  }
  k_pack_b<<<dim3((unsigned)((G * IN + 255) / 256)), blk, 0, stream>>>(Wih0, W0ihF, (int)IN, 0, (int)G, 1);
  k_pack_b<<<dim3((unsigned)((G * P + 255) / 256)),  blk, 0, stream>>>(Whh0, W0hhF, (int)P, 0, (int)G, 1);
  k_pack_b<<<dim3((unsigned)((G * P + 255) / 256)),  blk, 0, stream>>>(Wih1, W1F,   (int)P, 0,      (int)G, 1);
  k_pack_b<<<dim3((unsigned)((G * P + 255) / 256)),  blk, 0, stream>>>(Whh1, W1F,   (int)P, (int)P, (int)G, 1);
  k_pack_b<<<dim3((unsigned)((P * H + 255) / 256)),  blk, 0, stream>>>(Whr0, Whr0F, (int)H, 0, (int)P, 0);
  k_pack_b<<<dim3((unsigned)((P * H + 255) / 256)),  blk, 0, stream>>>(Whr1, Whr1F, (int)H, 0, (int)P, 0);
  k_pack_bias<<<dim3(32), blk, 0, stream>>>(bih0, bhh0, b0);
  k_pack_bias<<<dim3(32), blk, 0, stream>>>(bih1, bhh1, b1);

  // zero initial state (c0, c1, h0, h1)
  (void)hipMemsetAsync(c0, 0, N * H * 4, stream);
  (void)hipMemsetAsync(c1, 0, N * H * 4, stream);
  (void)hipMemsetAsync(hcat, 0, N * 2 * P * 2, stream);

  const dim3 gGate(128, 32);  // 8192/64 x 8192/256
  const dim3 gProj(4, 32);    // 256/64  x 8192/256

  for (int t = 0; t < 16; ++t) {
    // layer 0 gates + cell update.  t==0: h0==0 so gates = x@Wih0^T + b0.
    if (t == 0) {
      k_lstm_wmma<0><<<gGate, blk, 0, stream>>>(xbf, (int)IN, (int)IN, W0ihF, (int)(G >> 4),
                                                b0, c0, pbuf, nullptr, 0, nullptr);
    } else {
      k_lstm_wmma<0><<<gGate, blk, 0, stream>>>(hcat, 512, (int)P, W0hhF, (int)(G >> 4),
                                                b0, c0, pbuf, nullptr, 0, nullptr);
    }
    // layer 0 projection: h0 = p0 @ Whr0^T -> hcat[:, 0:256]
    k_lstm_wmma<1><<<gProj, blk, 0, stream>>>(pbuf, (int)H, (int)H, Whr0F, (int)(P >> 4),
                                              nullptr, nullptr, nullptr, hcat, 0, nullptr);
    // layer 1 gates: [h0|h1] @ [Wih1;Whh1]^T + b1  (h1 half is zero at t=0)
    k_lstm_wmma<0><<<gGate, blk, 0, stream>>>(hcat, 512, 512, W1F, (int)(G >> 4),
                                              b1, c1, pbuf, nullptr, 0, nullptr);
    // layer 1 projection: h1 -> hcat[:, 256:512] and hseq[t]
    k_lstm_wmma<1><<<gProj, blk, 0, stream>>>(pbuf, (int)H, (int)H, Whr1F, (int)(P >> 4),
                                              nullptr, nullptr, nullptr, hcat, 256,
                                              hseq + (size_t)t * N * P);
  }

  // softmax over P=256 with (t, n) -> (b, bar*16+t) remap; f32 output.
  k_softmax<<<dim3(16384), blk, 0, stream>>>(hseq, (float*)d_out);
}
